// ActionHead_46411416600827
// MI455X (gfx1250) — matrix operand order repaired and checked by reading, not compiled, not run on hardware
//
#include <hip/hip_runtime.h>

typedef __attribute__((ext_vector_type(16))) __bf16 v16bf;
typedef __attribute__((ext_vector_type(8)))  __bf16 v8bf;
typedef __attribute__((ext_vector_type(2)))  __bf16 v2bf;
typedef __attribute__((ext_vector_type(8)))  float  v8f;

#define DIM      128
#define HID      128
#define KTOT     257          // 2*DIM + 1
#define KS       296          // LDS row stride (elements), padded for banks
#define MTILE    16
#define MT_PER_WAVE 2         // each wave: 2 M-tiles -> 32 actions, B frag reused 2x
#define WAVES    8
#define BLOCK_ROWS (MTILE * MT_PER_WAVE * WAVES)   // 256 actions per block
#define NCHUNKS  9            // 9 * 32 = 288 >= 257 (chunk 8 = soldiers tail)

__global__ __launch_bounds__(256) void action_head_fused(
    const float* __restrict__ emb,      // [50000][128]
    const float* __restrict__ skipA,    // [128]
    const float* __restrict__ skipD,    // [128]
    const float* __restrict__ W1,       // [257][128]
    const float* __restrict__ B1,       // [128]
    const float* __restrict__ W2,       // [128]
    const float* __restrict__ B2,       // [1]
    const int*   __restrict__ lut,      // [N][3]
    float* __restrict__ out,            // [N]
    int n_actions)
{
  extern __shared__ __bf16 w1t[];   // [HID][KS] : W1 transposed, bf16, zero-padded K

  const int tid = threadIdx.x;

  // ---- Stage W1^T into LDS: paired k -> one cvt_pk + one b32 store ----
  for (int idx = tid; idx < HID * (KS / 2); idx += 256) {
    const int n  = idx & (HID - 1);     // 0..127 (fast-varying -> coalesced)
    const int k  = (idx >> 7) * 2;      // even k
    float v0 = (k     < KTOT) ? W1[k * HID + n]       : 0.0f;
    float v1 = (k + 1 < KTOT) ? W1[(k + 1) * HID + n] : 0.0f;
    v2bf pk; pk[0] = (__bf16)v0; pk[1] = (__bf16)v1;
    *(v2bf*)(w1t + n * KS + k) = pk;    // 4B aligned (k even, KS even)
  }
  __syncthreads();

  const int lane = tid & 31;
  const int wave = tid >> 5;
  const int nloc = lane & 15;          // A: row M within tile; B/C: column within tile
  const int half = lane >> 4;          // K-half selector per ISA layouts

  const int row_base0 = (int)blockIdx.x * BLOCK_ROWS + wave * (MTILE * MT_PER_WAVE);
  const int row_base1 = row_base0 + MTILE;

  // Per-lane gather pointers for the two M-tiles (lane supplies row m == nloc)
  const float* aptr[2]; const float* dptr[2]; float soldiers[2];
  {
    int rb[2] = { row_base0, row_base1 };
    #pragma unroll
    for (int t = 0; t < 2; ++t) {
      int arow   = rb[t] + nloc;
      int arow_c = (arow < n_actions) ? arow : (n_actions - 1);
      int atk = lut[3 * arow_c + 0];
      int dfd = lut[3 * arow_c + 1];
      soldiers[t] = (float)lut[3 * arow_c + 2];
      bool skip = (atk == -1);
      aptr[t] = skip ? skipA : (emb + (long long)(atk < 0 ? 0 : atk) * DIM);
      dptr[t] = skip ? skipD : (emb + (long long)dfd * DIM);
    }
  }

  // Lane-fixed part of the B-fragment address (nt/c contribute uniform offsets)
  const __bf16* bbase = w1t + nloc * KS + half * 8;
  #define LOAD_B(c_, nt_)                                                     \
    ({ const __bf16* bp_ = bbase + (nt_) * (16 * KS) + (c_) * 32;             \
       v8bf lo_ = *(const v8bf*)(bp_);                                        \
       v8bf hi_ = *(const v8bf*)(bp_ + 16);                                   \
       __builtin_shufflevector(lo_, hi_, 0,1,2,3,4,5,6,7,8,9,10,11,12,13,14,15); })

  // 16 accumulator tiles: [tile][nt] ; n = nt*16 + nloc, m = r + half*8
  v8f acc0[8], acc1[8];
  #pragma unroll
  for (int nt = 0; nt < 8; ++nt) {
    #pragma unroll
    for (int r = 0; r < 8; ++r) { acc0[nt][r] = 0.0f; acc1[nt][r] = 0.0f; }
  }

  // ---- Unified K loop over 9 chunks, B fragments software-pipelined ----
  v16bf bcur = LOAD_B(0, 0);   // prime the pipeline

  #pragma unroll
  for (int c = 0; c < NCHUNKS; ++c) {
    // Build the two A fragments for this chunk (ISA 16-bit A layout)
    v16bf a0, a1;
    if (c < 8) {
      const int kb = (c < 4) ? (32 * c) : (32 * c - DIM);
      #pragma unroll
      for (int t = 0; t < 2; ++t) {
        const float* src = (c < 4) ? aptr[t] : dptr[t];
        const float4 f0 = *(const float4*)(src + kb + half * 8);
        const float4 f1 = *(const float4*)(src + kb + half * 8 + 4);
        const float4 f2 = *(const float4*)(src + kb + 16 + half * 8);
        const float4 f3 = *(const float4*)(src + kb + 16 + half * 8 + 4);
        v16bf a;
        a[0]  = (__bf16)f0.x; a[1]  = (__bf16)f0.y; a[2]  = (__bf16)f0.z; a[3]  = (__bf16)f0.w;
        a[4]  = (__bf16)f1.x; a[5]  = (__bf16)f1.y; a[6]  = (__bf16)f1.z; a[7]  = (__bf16)f1.w;
        a[8]  = (__bf16)f2.x; a[9]  = (__bf16)f2.y; a[10] = (__bf16)f2.z; a[11] = (__bf16)f2.w;
        a[12] = (__bf16)f3.x; a[13] = (__bf16)f3.y; a[14] = (__bf16)f3.z; a[15] = (__bf16)f3.w;
        if (t == 0) a0 = a; else a1 = a;
      }
    } else {
      // Tail chunk: only K=256 (n_soldiers) non-zero -> half 0, elem 0
      #pragma unroll
      for (int i = 0; i < 16; ++i) { a0[i] = (__bf16)0.0f; a1[i] = (__bf16)0.0f; }
      a0[0] = (__bf16)((half == 0) ? soldiers[0] : 0.0f);
      a1[0] = (__bf16)((half == 0) ? soldiers[1] : 0.0f);
    }

    #pragma unroll
    for (int nt = 0; nt < 8; ++nt) {
      // Prefetch the NEXT B fragment (crosses chunk boundaries) so the current
      // WMMA pair only needs dscnt <= outstanding-next, not dscnt==0.
      v16bf bnext = bcur;
      const int i = c * 8 + nt + 1;          // flattened next index
      if (i < NCHUNKS * 8) bnext = LOAD_B(i >> 3, i & 7);

      acc0[nt] = __builtin_amdgcn_wmma_f32_16x16x32_bf16(
          false, a0, false, bcur, (short)0, acc0[nt], false, false);
      acc1[nt] = __builtin_amdgcn_wmma_f32_16x16x32_bf16(
          false, a1, false, bcur, (short)0, acc1[nt], false, false);
      bcur = bnext;
    }
  }
  #undef LOAD_B

  // ---- Fused second layer: relu(h)·W2, reduced across the 16 N-lanes ----
  const float b2v = B2[0];
  float p0[8], p1[8];
  #pragma unroll
  for (int r = 0; r < 8; ++r) { p0[r] = 0.0f; p1[r] = 0.0f; }

  #pragma unroll
  for (int nt = 0; nt < 8; ++nt) {
    const float b1v = B1[nt * 16 + nloc];
    const float w2v = W2[nt * 16 + nloc];
    #pragma unroll
    for (int r = 0; r < 8; ++r) {
      float h0 = acc0[nt][r] + b1v; h0 = (h0 > 0.0f) ? h0 : 0.0f;
      float h1 = acc1[nt][r] + b1v; h1 = (h1 > 0.0f) ? h1 : 0.0f;
      p0[r] = fmaf(h0, w2v, p0[r]);
      p1[r] = fmaf(h1, w2v, p1[r]);
    }
  }

  // butterfly sum within each 16-lane half (masks < 16 stay inside the half)
  #pragma unroll
  for (int r = 0; r < 8; ++r) {
    p0[r] += __shfl_xor(p0[r], 1, 32);
    p0[r] += __shfl_xor(p0[r], 2, 32);
    p0[r] += __shfl_xor(p0[r], 4, 32);
    p0[r] += __shfl_xor(p0[r], 8, 32);
    p1[r] += __shfl_xor(p1[r], 1, 32);
    p1[r] += __shfl_xor(p1[r], 2, 32);
    p1[r] += __shfl_xor(p1[r], 4, 32);
    p1[r] += __shfl_xor(p1[r], 8, 32);
  }

  if (nloc == 0) {   // lanes 0 (m = r) and 16 (m = r + 8) store
    #pragma unroll
    for (int r = 0; r < 8; ++r) {
      int row0 = row_base0 + r + half * 8;
      int row1 = row_base1 + r + half * 8;
      if (row0 < n_actions) out[row0] = p0[r] + b2v;
      if (row1 < n_actions) out[row1] = p1[r] + b2v;
    }
  }
}

extern "C" void kernel_launch(void* const* d_in, const int* in_sizes, int n_in,
                              void* d_out, int out_size, void* d_ws, size_t ws_size,
                              hipStream_t stream) {
  const float* emb   = (const float*)d_in[0];
  const float* skipA = (const float*)d_in[1];
  const float* skipD = (const float*)d_in[2];
  const float* W1    = (const float*)d_in[3];
  const float* B1    = (const float*)d_in[4];
  const float* W2    = (const float*)d_in[5];
  const float* B2    = (const float*)d_in[6];
  const int*   lut   = (const int*)d_in[7];
  float* out = (float*)d_out;

  const int n_actions = in_sizes[7] / 3;
  const int blocks = (n_actions + BLOCK_ROWS - 1) / BLOCK_ROWS;
  const size_t smem = (size_t)HID * KS * sizeof(__bf16);  // ~75.8 KB < 320 KB/WGP

  action_head_fused<<<blocks, 256, smem, stream>>>(
      emb, skipA, skipD, W1, B1, W2, B2, lut, out, n_actions);
}
